// GatedGraphBasic_17205638988467
// MI455X (gfx1250) — matrix operand order repaired and checked by reading, not compiled, not run on hardware
//
#include <hip/hip_runtime.h>

#define BB      4
#define NSLICE  128
#define NTYPE   8
#define NPS     128
#define HDIM    64
#define NN      (NSLICE * NPS)

// padded LDS row strides (halfs): odd multiples of 8 -> 16B-aligned rows,
// lane start banks all distinct for 16-lane strided fragment loads
#define LD_HID  136   // sHidT rows of 128
#define LD_E    40    // sEt   rows of 32
#define LD_W    72    // sWT   rows of 64
#define LD_ACT  72    // sAct  rows of 64

// LDS carve (halfs)
#define OFF_HID 0
#define OFF_E   (OFF_HID + HDIM * LD_HID)       // 8704
#define OFF_W   (OFF_E   + NPS * LD_E)          // 13824
#define OFF_ACT (OFF_W   + 3 * HDIM * LD_W)     // 27648
#define SMEM_HALFS (OFF_ACT + NPS * LD_ACT)     // 36864
#define SMEM_BYTES (SMEM_HALFS * 2)             // 73728

typedef __attribute__((ext_vector_type(16))) _Float16 v16h;
typedef __attribute__((ext_vector_type(8)))  _Float16 h8;
typedef __attribute__((ext_vector_type(4)))  _Float16 h4;
typedef __attribute__((ext_vector_type(2)))  _Float16 h2;
typedef __attribute__((ext_vector_type(8)))  float    v8f;

// packed f32x2 -> f16x2 convert (v_cvt_pk_rtz_f16_f32); bit-cast fixes the
// __fp16-vs-_Float16 vector type mismatch of the builtin's return type.
__device__ __forceinline__ h2 cvt_pk(float x, float y) {
  return __builtin_bit_cast(h2, __builtin_amdgcn_cvt_pkrtz(x, y));
}

// ---------------- WMMA fragment loads (all 16B ds_load_b128) ----------------

// A fragment from row-major M x K storage: A[m,k] = M[m*ld + k].
__device__ __forceinline__ v16h frag_a(const _Float16* M, int ld, int mBase, int kBase, int lane) {
  const _Float16* p = M + (mBase + (lane & 15)) * ld + kBase + ((lane >> 4) << 4);
  h8 lo = *(const h8*)p;          // fragment elements 0..7  (K +0..7)
  h8 hi = *(const h8*)(p + 8);    // fragment elements 8..15 (K +8..15)
  return __builtin_shufflevector(lo, hi, 0, 1, 2, 3, 4, 5, 6, 7, 8, 9, 10, 11, 12, 13, 14, 15);
}

// B fragment from TRANSPOSED storage: B[k,n] stored as Mt[n*ld + k].
__device__ __forceinline__ v16h frag_bT(const _Float16* Mt, int ld, int kBase, int nBase, int lane) {
  const _Float16* p = Mt + (nBase + (lane & 15)) * ld + kBase + ((lane >> 4) << 4);
  h8 lo = *(const h8*)p;
  h8 hi = *(const h8*)(p + 8);
  return __builtin_shufflevector(lo, hi, 0, 1, 2, 3, 4, 5, 6, 7, 8, 9, 10, 11, 12, 13, 14, 15);
}

// ---------------- cooperative staging ----------------

// row-major K x N f32 -> transposed N x K f16 (row stride ldl), 4x4 micro-tiles:
// 4 coalesced global b128 loads + 8 packed cvt + 4 contiguous ds b64 stores.
__device__ __forceinline__ void stage_T(const float* __restrict__ g, _Float16* l,
                                        int K, int N, int ldl, int tid, int nt) {
  int ntile = N >> 2;
  int tiles = (K >> 2) * ntile;
  for (int i = tid; i < tiles; i += nt) {
    int kt = i / ntile;
    int kb = kt << 2;
    int nb = (i - kt * ntile) << 2;
    const float* gp = g + kb * N + nb;
    float4 r0 = *(const float4*)(gp);
    float4 r1 = *(const float4*)(gp + N);
    float4 r2 = *(const float4*)(gp + 2 * N);
    float4 r3 = *(const float4*)(gp + 3 * N);
    _Float16* lp = l + nb * ldl + kb;
    h2 a, b;
    a = cvt_pk(r0.x, r1.x); b = cvt_pk(r2.x, r3.x);
    *(h4*)(lp)           = __builtin_shufflevector(a, b, 0, 1, 2, 3);
    a = cvt_pk(r0.y, r1.y); b = cvt_pk(r2.y, r3.y);
    *(h4*)(lp + ldl)     = __builtin_shufflevector(a, b, 0, 1, 2, 3);
    a = cvt_pk(r0.z, r1.z); b = cvt_pk(r2.z, r3.z);
    *(h4*)(lp + 2 * ldl) = __builtin_shufflevector(a, b, 0, 1, 2, 3);
    a = cvt_pk(r0.w, r1.w); b = cvt_pk(r2.w, r3.w);
    *(h4*)(lp + 3 * ldl) = __builtin_shufflevector(a, b, 0, 1, 2, 3);
  }
}

// row-major f32 -> row-major f16 with padded row stride
__device__ __forceinline__ void stage_rm(const float* __restrict__ g, _Float16* l,
                                         int N, int rows, int ldl, int tid, int nt) {
  int n4 = rows * N / 4;
  const float4* g4 = (const float4*)g;
  for (int i = tid; i < n4; i += nt) {
    float4 v = g4[i];
    int r = (4 * i) / N;
    int c = (4 * i) % N;
    h2 a = cvt_pk(v.x, v.y);
    h2 b = cvt_pk(v.z, v.w);
    *(h4*)(l + r * ldl + c) = __builtin_shufflevector(a, b, 0, 1, 2, 3);
  }
}

__device__ __forceinline__ float fast_sigmoid(float x) {
  return __fdividef(1.f, 1.f + __expf(-x));
}
__device__ __forceinline__ float fast_tanh(float x) {
  float e = __expf(-2.f * x);
  return __fdividef(1.f - e, 1.f + e);
}

__global__ __launch_bounds__(256) void ggnn_fused(
    const float* __restrict__ hidden,   // [B, N, H]
    const float* __restrict__ edge,     // [B, NSLICE, NTYPE, NPS, NPS]
    const float* __restrict__ ba,       // [NTYPE, H]
    const float* __restrict__ wzrh,     // [NTYPE, H, 3H]
    const float* __restrict__ uzur,     // [H, 2H]
    const float* __restrict__ uh,       // [H, H]
    float* __restrict__ out)            // [B, N, H]
{
  extern __shared__ __align__(16) _Float16 smem[];
  _Float16* sHidT = smem + OFF_HID;   // hidden^T [64 h][ld 136]  (B op, M1)
  _Float16* sEt   = smem + OFF_E;     // E-chunk^T [128 m][ld 40] (A op, M1); reused: uh^T
  _Float16* sWT   = smem + OFF_W;     // W_t^T [192 k'][ld 72]    (B op, M2); reused: uzur^T
  _Float16* sAct  = smem + OFF_ACT;   // act_t [128 m][ld 72]     (A op, M2); reused: hid rm, r*h

  const int tid   = threadIdx.x;
  const int lane  = tid & 31;
  const int wave  = tid >> 5;          // 0..7, each owns 16 output rows
  const int mBase = wave * 16;
  const int b = blockIdx.x / NSLICE;
  const int s = blockIdx.x % NSLICE;

  const float* hidG = hidden + ((size_t)b * NN + (size_t)s * NPS) * HDIM;
  float*       outG = out    + ((size_t)b * NN + (size_t)s * NPS) * HDIM;

  // hidden^T once per block: [128 n][64 h] -> sHidT[h][n]
  stage_T(hidG, sHidT, NPS, HDIM, LD_HID, tid, 256);

  v8f accW[12];                        // wzrh rows mBase..+15, cols 0..191
#pragma unroll
  for (int j = 0; j < 12; ++j)
#pragma unroll
    for (int v = 0; v < 8; ++v) accW[j][v] = 0.f;

  for (int t = 0; t < NTYPE; ++t) {
    __syncthreads();                   // prior reads of sWT / sAct done
    // W_t [64 h][192 k'] -> sWT[k'][h]
    stage_T(wzrh + (size_t)t * HDIM * 3 * HDIM, sWT, HDIM, 3 * HDIM, LD_W, tid, 256);

    v8f accA[4];
#pragma unroll
    for (int j = 0; j < 4; ++j)
#pragma unroll
      for (int v = 0; v < 8; ++v) accA[j][v] = 0.f;

    const float* eG = edge + ((((size_t)b * NSLICE + s) * NTYPE + t) * NPS) * NPS;

    // M1: act = E^T (128x128) @ H (128x64), K chunked by 32
    for (int kk = 0; kk < 4; ++kk) {
      __syncthreads();                 // previous chunk consumed (publishes sWT on kk==0)
      // E chunk [32 n][128 m] -> sEt[m][n_local]
      stage_T(eG + (size_t)kk * 32 * NPS, sEt, 32, NPS, LD_E, tid, 256);
      if (kk < 3) __builtin_prefetch(eG + (size_t)(kk + 1) * 32 * NPS + tid * 16, 0, 0);
      __syncthreads();
      v16h a = frag_a(sEt, LD_E, mBase, 0, lane);
#pragma unroll
      for (int j = 0; j < 4; ++j) {
        v16h bf = frag_bT(sHidT, LD_HID, kk * 32, j * 16, lane);
        accA[j] = __builtin_amdgcn_wmma_f32_16x16x32_f16(
            false, a, false, bf, (short)0, accA[j], false, false);
      }
    }

    // +ba[t], convert to f16, park in sAct (each wave writes only its own rows)
#pragma unroll
    for (int j = 0; j < 4; ++j) {
      int   n   = j * 16 + (lane & 15);
      float bav = ba[t * HDIM + n];
      int   m0  = mBase + ((lane >> 4) << 3);
#pragma unroll
      for (int v = 0; v < 8; ++v)
        sAct[(m0 + v) * LD_ACT + n] = (_Float16)(accA[j][v] + bav);
    }

    // M2: wzrh += act (128x64) @ W_t (64x192)
#pragma unroll
    for (int kk = 0; kk < 2; ++kk) {
      v16h a2 = frag_a(sAct, LD_ACT, mBase, kk * 32, lane);
#pragma unroll
      for (int j = 0; j < 12; ++j) {
        v16h b2 = frag_bT(sWT, LD_W, kk * 32, j * 16, lane);
        accW[j] = __builtin_amdgcn_wmma_f32_16x16x32_f16(
            false, a2, false, b2, (short)0, accW[j], false, false);
      }
    }
  }

  // ---------------- GRU phase ----------------
  __syncthreads();                     // sWT / sEt / sAct free now
  _Float16* sUzurT = sWT;              // uzur^T: [128 k'][ld 72]
  _Float16* sUhT   = sEt;              // uh^T:   [64 k'][ld 72] (4608 <= 5120 halfs)
  stage_T(uzur, sUzurT, HDIM, 2 * HDIM, LD_W, tid, 256);
  stage_T(uh,   sUhT,   HDIM, HDIM,     LD_ACT, tid, 256);
  stage_rm(hidG, sAct, HDIM, NPS, LD_ACT, tid, 256);   // hidden row-major f16 (A operand)
  __syncthreads();

  // hidden @ uz_ur : rows mBase..+15, 128 cols (tiles 0..3 = uz, 4..7 = ur)
  v8f accU[8];
#pragma unroll
  for (int j = 0; j < 8; ++j)
#pragma unroll
    for (int v = 0; v < 8; ++v) accU[j][v] = 0.f;
#pragma unroll
  for (int kk = 0; kk < 2; ++kk) {
    v16h a = frag_a(sAct, LD_ACT, mBase, kk * 32, lane);
#pragma unroll
    for (int j = 0; j < 8; ++j) {
      v16h bf = frag_bT(sUzurT, LD_W, kk * 32, j * 16, lane);
      accU[j] = __builtin_amdgcn_wmma_f32_16x16x32_f16(
          false, a, false, bf, (short)0, accU[j], false, false);
    }
  }

  // r = sigmoid(wr + ur);  r*h (f32 hidden from global) -> sAct (own rows)
  {
    int n15 = lane & 15;
    int m0  = mBase + ((lane >> 4) << 3);
#pragma unroll
    for (int j = 0; j < 4; ++j) {
      int hcol = j * 16 + n15;
#pragma unroll
      for (int v = 0; v < 8; ++v) {
        float rr = fast_sigmoid(accW[4 + j][v] + accU[4 + j][v]);
        float hv = hidG[(m0 + v) * HDIM + hcol];
        sAct[(m0 + v) * LD_ACT + hcol] = (_Float16)(rr * hv);
      }
    }
  }

  // (r*h) @ uh : rows mBase..+15, 64 cols
  v8f accT[4];
#pragma unroll
  for (int j = 0; j < 4; ++j)
#pragma unroll
    for (int v = 0; v < 8; ++v) accT[j][v] = 0.f;
#pragma unroll
  for (int kk = 0; kk < 2; ++kk) {
    v16h a = frag_a(sAct, LD_ACT, mBase, kk * 32, lane);
#pragma unroll
    for (int j = 0; j < 4; ++j) {
      v16h bf = frag_bT(sUhT, LD_ACT, kk * 32, j * 16, lane);
      accT[j] = __builtin_amdgcn_wmma_f32_16x16x32_f16(
          false, a, false, bf, (short)0, accT[j], false, false);
    }
  }

  // out = (1-z)*h + z*tanh(wh + rhUh),  z = sigmoid(wz + uz)
  {
    int n15 = lane & 15;
    int m0  = mBase + ((lane >> 4) << 3);
#pragma unroll
    for (int j = 0; j < 4; ++j) {
      int hcol = j * 16 + n15;
#pragma unroll
      for (int v = 0; v < 8; ++v) {
        float z  = fast_sigmoid(accW[j][v] + accU[j][v]);
        float ht = fast_tanh(accW[8 + j][v] + accT[j][v]);
        float hv = hidG[(m0 + v) * HDIM + hcol];
        outG[(m0 + v) * HDIM + hcol] = (1.f - z) * hv + z * ht;
      }
    }
  }
}

extern "C" void kernel_launch(void* const* d_in, const int* in_sizes, int n_in,
                              void* d_out, int out_size, void* d_ws, size_t ws_size,
                              hipStream_t stream) {
  const float* hidden = (const float*)d_in[0];
  const float* edge   = (const float*)d_in[1];
  const float* ba     = (const float*)d_in[2];
  const float* wzrh   = (const float*)d_in[3];
  const float* uzur   = (const float*)d_in[4];
  const float* uh     = (const float*)d_in[5];
  float* out = (float*)d_out;

  dim3 grid(BB * NSLICE);   // 512 workgroups, one per (batch, slice) 128-note block
  dim3 block(256);          // 8 wave32 waves
  hipLaunchKernelGGL(ggnn_fused, grid, block, SMEM_BYTES, stream,
                     hidden, edge, ba, wzrh, uzur, uh, out);
}
